// GCNNetBase_56719338111685
// MI455X (gfx1250) — compile-verified, matrix operand order checked by reading, not run on hardware
//
#include <hip/hip_runtime.h>

// ---------------------------------------------------------------------------
// GCN (3x GraphConv + BN + ReLU, mean-pool, graph-BN, 2 linear heads)
// MI455X / gfx1250: wave32, WMMA bf16 (hi/lo split for ~fp32 GEMM precision),
// async global->LDS staging of weight fragments (ASYNCcnt path),
// hardware f32 global atomics for the edge scatter (L2-resident buffers).
// ---------------------------------------------------------------------------

constexpr int DIM  = 128;   // feature dim
constexpr int GNUM = 128;   // number of graphs
constexpr float EPSV = 1e-5f;

typedef __attribute__((ext_vector_type(16))) __bf16 v16bf;
typedef __attribute__((ext_vector_type(8)))  float  v8f;

// ---- helpers ---------------------------------------------------------------

__device__ __forceinline__ unsigned short f2bf_rn(float f) {
  unsigned int u = __float_as_uint(f);
  unsigned int r = u + 0x7fffu + ((u >> 16) & 1u);
  return (unsigned short)(r >> 16);
}

// split fp32 into bf16 hi (truncation, exact) + bf16 lo (rounded remainder)
__device__ __forceinline__ void split_bf(float v, unsigned short& h, unsigned short& l) {
  unsigned int u = __float_as_uint(v);
  h = (unsigned short)(u >> 16);
  float fh = __uint_as_float((unsigned int)h << 16);
  l = f2bf_rn(v - fh);
}

// fragment K index for 16-bit A/B layout (16x32 tile), element j of v16bf
__device__ __forceinline__ int frag_k(int ktile, int hoff, int j) {
  return (ktile << 5) + hoff + j + ((j >= 8) ? 8 : 0);
}

// gfx1250 async global->LDS copy, GVS form: lds[vdst] = mem[saddr + vaddr]
__device__ __forceinline__ void async_copy_b128(unsigned int lds_byte_addr,
                                                unsigned long long gbase,
                                                unsigned int gbyte_off) {
  asm volatile("global_load_async_to_lds_b128 %0, %1, %2"
               :: "v"(lds_byte_addr), "v"(gbyte_off), "s"(gbase)
               : "memory");
}

__device__ __forceinline__ void wait_async0() {
  asm volatile("s_wait_asynccnt 0x0" ::: "memory");
}

// ---- generic zero ----------------------------------------------------------

__global__ void k_zero(float* __restrict__ p, int n) {
  int i = blockIdx.x * blockDim.x + threadIdx.x;
  if (i < n) p[i] = 0.0f;
}

// ---- degrees & norms -------------------------------------------------------

__global__ void k_edge_deg(const int* __restrict__ src, const int* __restrict__ dst,
                           float* __restrict__ dout, float* __restrict__ din, int E) {
  int e = blockIdx.x * blockDim.x + threadIdx.x;
  if (e >= E) return;
  unsafeAtomicAdd(&dout[src[e]], 1.0f);
  unsafeAtomicAdd(&din[dst[e]], 1.0f);
}

__global__ void k_norms(float* __restrict__ dout, float* __restrict__ din, int N) {
  int i = blockIdx.x * blockDim.x + threadIdx.x;
  if (i >= N) return;
  float a = dout[i]; dout[i] = rsqrtf(a < 1.0f ? 1.0f : a);
  float b = din[i];  din[i]  = rsqrtf(b < 1.0f ? 1.0f : b);
}

// ---- pack X*(norm_out) into A-fragment layout (hi/lo bf16) -----------------
// word index: [mtile][ktile(4)][lane(32)][w(8)]  (one u32 = 2 bf16, K even|odd)

__global__ void k_pack_x(const float* __restrict__ x, const float* __restrict__ norm,
                         unsigned int* __restrict__ fh, unsigned int* __restrict__ fl,
                         int mtiles, int N) {
  int idx = blockIdx.x * blockDim.x + threadIdx.x;
  if (idx >= mtiles * 1024) return;
  int mtile = idx >> 10;
  int rem   = idx & 1023;
  int kt    = rem >> 8;
  int lane  = (rem >> 3) & 31;
  int w     = rem & 7;
  int m     = (mtile << 4) + (lane & 15);
  int hoff  = (lane >> 4) << 3;      // lanes 16-31 hold K base +8
  int j0    = w << 1;
  int k0    = frag_k(kt, hoff, j0);  // j0 even -> k1 = k0+1 in same region
  float v0 = 0.0f, v1 = 0.0f;
  if (m < N) {
    float s = norm[m];
    const float* row = x + (size_t)m * DIM;
    v0 = row[k0] * s;
    v1 = row[k0 + 1] * s;
  }
  unsigned short h0, l0, h1, l1;
  split_bf(v0, h0, l0);
  split_bf(v1, h1, l1);
  fh[idx] = ((unsigned int)h1 << 16) | h0;
  fl[idx] = ((unsigned int)l1 << 16) | l0;
}

// ---- pack W into B-fragment layout (hi/lo bf16) ----------------------------
// word index: [ktile(4)][ntile(8)][lane(32)][w(8)]   (8192 words)

__global__ void k_pack_w(const float* __restrict__ W,
                         unsigned int* __restrict__ wh, unsigned int* __restrict__ wl) {
  int idx = blockIdx.x * blockDim.x + threadIdx.x;
  if (idx >= 8192) return;
  int kt   = idx >> 11;
  int rem  = idx & 2047;
  int nt   = rem >> 8;
  int lane = (rem >> 3) & 31;
  int w    = rem & 7;
  int n    = (nt << 4) + (lane & 15);
  int hoff = (lane >> 4) << 3;
  int j0   = w << 1;
  int k0   = frag_k(kt, hoff, j0);
  float v0 = W[(size_t)k0 * DIM + n];
  float v1 = W[(size_t)(k0 + 1) * DIM + n];
  unsigned short h0, l0, h1, l1;
  split_bf(v0, h0, l0);
  split_bf(v1, h1, l1);
  wh[idx] = ((unsigned int)h1 << 16) | h0;
  wl[idx] = ((unsigned int)l1 << 16) | l0;
}

// ---- WMMA GEMM: h[N,128] = Xs[N,128] @ W[128,128] --------------------------
// Block: async-stage the 64KB of B fragments (hi+lo) into LDS once, then each
// wave computes one 16-row stripe x all 8 N-tiles; 4 K-tiles x 3 WMMAs.

__global__ void __launch_bounds__(256)
k_gemm(const unsigned int* __restrict__ aHi, const unsigned int* __restrict__ aLo,
       const unsigned int* __restrict__ bHi, const unsigned int* __restrict__ bLo,
       float* __restrict__ out, int mtiles, int N) {
  __shared__ unsigned int sW[16384];    // [0..8191]=hi, [8192..16383]=lo (64KB)

  // --- async global->LDS staging (all 256 threads participate) ---
  {
    int t = threadIdx.x;
    unsigned int ldsbase = (unsigned int)(size_t)(void*)sW;
    unsigned long long gHi = (unsigned long long)(size_t)bHi;
    unsigned long long gLo = (unsigned long long)(size_t)bLo;
#pragma unroll
    for (int i = 0; i < 8; ++i) {
      unsigned int boff = (unsigned int)(i * 256 + t) * 16u;  // b128 chunks
      async_copy_b128(ldsbase + boff, gHi, boff);
      async_copy_b128(ldsbase + 32768u + boff, gLo, boff);
    }
    wait_async0();
  }
  __syncthreads();

  int wave = threadIdx.x >> 5;
  int lane = threadIdx.x & 31;
  int mtile = blockIdx.x * 8 + wave;

  if (mtile < mtiles) {                 // wave-uniform: EXEC all-ones for WMMA
    const unsigned int* sHi = sW;
    const unsigned int* sLo = sW + 8192;

    v8f acc[8] = {};
    for (int kt = 0; kt < 4; ++kt) {
      int abase = ((mtile * 4 + kt) * 32 + lane) * 8;
      v16bf ah = *reinterpret_cast<const v16bf*>(aHi + abase);
      v16bf al = *reinterpret_cast<const v16bf*>(aLo + abase);
#pragma unroll
      for (int nt = 0; nt < 8; ++nt) {
        int bbase = ((kt * 8 + nt) * 32 + lane) * 8;
        v16bf bh = *reinterpret_cast<const v16bf*>(sHi + bbase);
        v16bf bl = *reinterpret_cast<const v16bf*>(sLo + bbase);
        acc[nt] = __builtin_amdgcn_wmma_f32_16x16x32_bf16(false, ah, false, bh,
                                                          (short)0, acc[nt], false, false);
        acc[nt] = __builtin_amdgcn_wmma_f32_16x16x32_bf16(false, ah, false, bl,
                                                          (short)0, acc[nt], false, false);
        acc[nt] = __builtin_amdgcn_wmma_f32_16x16x32_bf16(false, al, false, bh,
                                                          (short)0, acc[nt], false, false);
      }
    }
    // C/D layout: lanes 0-15 -> M=r, lanes 16-31 -> M=r+8; N = lane&15
    int mrow = (mtile << 4) + ((lane >> 4) << 3);
    int ncol = lane & 15;
#pragma unroll
    for (int nt = 0; nt < 8; ++nt) {
      float* o = out + (size_t)mrow * DIM + nt * 16 + ncol;
#pragma unroll
      for (int r = 0; r < 8; ++r) {
        if (mrow + r < N) o[(size_t)r * DIM] = acc[nt][r];
      }
    }
  }
}

// ---- edge scatter: agg[dst] += h[src]  (float4 + 4 hw f32 atomics) ---------

__global__ void k_scatter(const int* __restrict__ src, const int* __restrict__ dst,
                          const float* __restrict__ h, float* __restrict__ agg, int E) {
  int idx = blockIdx.x * blockDim.x + threadIdx.x;
  if (idx >= E * 32) return;
  int e = idx >> 5;
  int c = (idx & 31) << 2;
  int s = src[e], t = dst[e];
  const float4 v = *reinterpret_cast<const float4*>(h + (size_t)s * DIM + c);
  float* a = agg + (size_t)t * DIM + c;
  unsafeAtomicAdd(a + 0, v.x);
  unsafeAtomicAdd(a + 1, v.y);
  unsafeAtomicAdd(a + 2, v.z);
  unsafeAtomicAdd(a + 3, v.w);
}

// ---- BN stats: h2 = agg*norm_in + b; accumulate per-feature sum/sumsq ------

__global__ void k_bn_stats(const float* __restrict__ agg, const float* __restrict__ norm_in,
                           const float* __restrict__ bias, float* __restrict__ h2,
                           float* __restrict__ stats, int N) {
  __shared__ float s1[256], s2[256];
  int t = threadIdx.x;
  int d = t & 127;
  int sub = t >> 7;
  int base = blockIdx.x * 64;
  float bd = bias[d];
  float bsum = 0.0f, bsq = 0.0f;
  for (int r = sub; r < 64; r += 2) {
    int n = base + r;
    if (n < N) {
      float v = agg[(size_t)n * DIM + d] * norm_in[n] + bd;
      h2[(size_t)n * DIM + d] = v;
      bsum += v; bsq += v * v;
    }
  }
  s1[t] = bsum; s2[t] = bsq;
  __syncthreads();
  if (t < 128) {
    unsafeAtomicAdd(&stats[d],       s1[t] + s1[t + 128]);
    unsafeAtomicAdd(&stats[128 + d], s2[t] + s2[t + 128]);
  }
}

// ---- BN apply + ReLU -------------------------------------------------------

__global__ void k_bn_apply(const float* __restrict__ h2, const float* __restrict__ stats,
                           const float* __restrict__ g, const float* __restrict__ be,
                           float* __restrict__ xout, int N) {
  int idx = blockIdx.x * blockDim.x + threadIdx.x;
  if (idx >= N * DIM) return;
  int d = idx & 127;
  float invN = 1.0f / (float)N;
  float mu  = stats[d] * invN;
  float var = stats[128 + d] * invN - mu * mu;
  float v = (h2[idx] - mu) * rsqrtf(var + EPSV) * g[d] + be[d];
  xout[idx] = v > 0.0f ? v : 0.0f;
}

// ---- graph mean-pool accumulation ------------------------------------------

__global__ void k_pool(const float* __restrict__ x, const int* __restrict__ gid,
                       float* __restrict__ gsum, float* __restrict__ gcnt, int N) {
  int idx = blockIdx.x * blockDim.x + threadIdx.x;
  if (idx >= N * DIM) return;
  int n = idx >> 7;
  int d = idx & 127;
  int gg = gid[n];
  unsafeAtomicAdd(&gsum[gg * DIM + d], x[idx]);
  if (d == 0) unsafeAtomicAdd(&gcnt[gg], 1.0f);
}

// ---- final: emb = gsum/cnt -> graph BN -> two heads (single block) ---------

__global__ void __launch_bounds__(128)
k_final(const float* __restrict__ gsum, const float* __restrict__ gcnt,
        const float* __restrict__ gn_g, const float* __restrict__ gn_b,
        const float* __restrict__ Wr, const float* __restrict__ br,
        const float* __restrict__ Wc, const float* __restrict__ bc,
        float* __restrict__ y, float* __restrict__ cl) {
  __shared__ float semb[GNUM * DIM];   // 64 KB (<= 320 KB WGP LDS)
  int t = threadIdx.x;                 // feature index 0..127
  float s1 = 0.0f, s2 = 0.0f;
  for (int g = 0; g < GNUM; ++g) {
    float cnt = gcnt[g];
    cnt = cnt < 1.0f ? 1.0f : cnt;
    float v = gsum[g * DIM + t] / cnt;
    semb[g * DIM + t] = v;
    s1 += v; s2 += v * v;
  }
  float mu  = s1 * (1.0f / GNUM);
  float var = s2 * (1.0f / GNUM) - mu * mu;
  float sc = rsqrtf(var + EPSV) * gn_g[t];
  float sh = gn_b[t] - mu * sc;
  for (int g = 0; g < GNUM; ++g)
    semb[g * DIM + t] = semb[g * DIM + t] * sc + sh;
  __syncthreads();
  // regression head: y[g] = emb[g,:] . Wr + br  (t == graph index here)
  float acc = br[0];
  for (int d = 0; d < DIM; ++d) acc += semb[t * DIM + d] * Wr[d];
  y[t] = acc;
  // concept head: [G,16]
  for (int c = 0; c < 16; ++c) {
    float a = bc[c];
    for (int d = 0; d < DIM; ++d) a += semb[t * DIM + d] * Wc[d * 16 + c];
    cl[t * 16 + c] = a;
  }
}

// ---------------------------------------------------------------------------

static inline int cdiv(long long a, int b) { return (int)((a + b - 1) / b); }

extern "C" void kernel_launch(void* const* d_in, const int* in_sizes, int n_in,
                              void* d_out, int out_size, void* d_ws, size_t ws_size,
                              hipStream_t stream) {
  const float* x    = (const float*)d_in[0];
  const int*   src  = (const int*)d_in[1];
  const int*   dst  = (const int*)d_in[2];
  const int*   gid  = (const int*)d_in[3];
  const float* W[3]  = {(const float*)d_in[5], (const float*)d_in[9],  (const float*)d_in[13]};
  const float* bb[3] = {(const float*)d_in[6], (const float*)d_in[10], (const float*)d_in[14]};
  const float* gm[3] = {(const float*)d_in[7], (const float*)d_in[11], (const float*)d_in[15]};
  const float* be[3] = {(const float*)d_in[8], (const float*)d_in[12], (const float*)d_in[16]};
  const float* gn_g = (const float*)d_in[17];
  const float* gn_b = (const float*)d_in[18];
  const float* Wr   = (const float*)d_in[19];
  const float* br   = (const float*)d_in[20];
  const float* Wc   = (const float*)d_in[21];
  const float* bc   = (const float*)d_in[22];

  const int N = in_sizes[0] / DIM;
  const int E = in_sizes[1];
  const int mtiles = (N + 15) / 16;

  // workspace carving (256B aligned)
  char* ws = (char*)d_ws;
  size_t off = 0;
  auto carve = [&](size_t bytes) -> void* {
    off = (off + 255) & ~(size_t)255;
    void* p = ws + off;
    off += bytes;
    return p;
  };
  float* deg_out = (float*)carve((size_t)N * 4);          // -> norm_out in place
  float* deg_in  = (float*)carve((size_t)N * 4);          // -> norm_in in place
  unsigned int* fragHi = (unsigned int*)carve((size_t)mtiles * 1024 * 4);
  unsigned int* fragLo = (unsigned int*)carve((size_t)mtiles * 1024 * 4);
  unsigned int* wHi    = (unsigned int*)carve(8192 * 4);
  unsigned int* wLo    = (unsigned int*)carve(8192 * 4);
  float* buf_h = (float*)carve((size_t)N * DIM * 4);
  float* agg   = (float*)carve((size_t)N * DIM * 4);
  float* xA    = (float*)carve((size_t)N * DIM * 4);
  float* xB    = (float*)carve((size_t)N * DIM * 4);
  float* stats = (float*)carve(256 * 4);
  float* gsum  = (float*)carve((size_t)GNUM * DIM * 4);
  float* gcnt  = (float*)carve((size_t)GNUM * 4);

  float* out_x = (float*)d_out;                 // [N,128]
  float* out_y = out_x + (size_t)N * DIM;       // [G,1]
  float* out_c = out_y + GNUM;                  // [G,16]

  const int B = 256;

  // degrees -> norms
  k_zero<<<cdiv(N, B), B, 0, stream>>>(deg_out, N);
  k_zero<<<cdiv(N, B), B, 0, stream>>>(deg_in, N);
  k_edge_deg<<<cdiv(E, B), B, 0, stream>>>(src, dst, deg_out, deg_in, E);
  k_norms<<<cdiv(N, B), B, 0, stream>>>(deg_out, deg_in, N);

  const float* xin[3]  = {x, xA, xB};
  float*       xout[3] = {xA, xB, out_x};

  for (int L = 0; L < 3; ++L) {
    k_pack_x<<<cdiv((long long)mtiles * 1024, B), B, 0, stream>>>(
        xin[L], deg_out, fragHi, fragLo, mtiles, N);
    k_pack_w<<<32, B, 0, stream>>>(W[L], wHi, wLo);
    k_zero<<<cdiv((long long)N * DIM, B), B, 0, stream>>>(agg, N * DIM);
    k_gemm<<<cdiv(mtiles, 8), B, 0, stream>>>(fragHi, fragLo, wHi, wLo, buf_h, mtiles, N);
    k_scatter<<<cdiv((long long)E * 32, B), B, 0, stream>>>(src, dst, buf_h, agg, E);
    k_zero<<<1, B, 0, stream>>>(stats, 256);
    k_bn_stats<<<cdiv(N, 64), B, 0, stream>>>(agg, deg_in, bb[L], buf_h, stats, N);
    k_bn_apply<<<cdiv((long long)N * DIM, B), B, 0, stream>>>(buf_h, stats, gm[L], be[L], xout[L], N);
  }

  // graph mean pooling + heads
  k_zero<<<cdiv(GNUM * DIM, B), B, 0, stream>>>(gsum, GNUM * DIM);
  k_zero<<<1, B, 0, stream>>>(gcnt, GNUM);
  k_pool<<<cdiv((long long)N * DIM, B), B, 0, stream>>>(out_x, gid, gsum, gcnt, N);
  k_final<<<1, 128, 0, stream>>>(gsum, gcnt, gn_g, gn_b, Wr, br, Wc, bc, out_y, out_c);
}
